// GATRepresentationNetwork_17806934409716
// MI455X (gfx1250) — matrix operand order repaired and checked by reading, not compiled, not run on hardware
//
#include <hip/hip_runtime.h>
#include <hip/hip_bf16.h>

// ---------------------------------------------------------------------------
// GAT representation network on MI455X (gfx1250), wave32 + WMMA f16.
// One wave = one graph (16 nodes). All GEMMs via v_wmma_f32_16x16x32_f16.
// ---------------------------------------------------------------------------

typedef __attribute__((ext_vector_type(16))) _Float16 v16h;
typedef __attribute__((ext_vector_type(8)))  _Float16 v8h;
typedef __attribute__((ext_vector_type(8)))  float    v8f;

// ---- geometry ----
constexpr int NN      = 16;    // nodes per graph
constexpr int F1      = 256;   // heads*hid
constexpr int HSTRIDE = 264;   // halves, row stride for [16 x 256] f16 LDS bufs
constexpr int WTSTRIDE = 24;   // halves, row stride for wh^T [256 x 16] LDS buf

// ---- workspace layout (in halves), weights pre-converted to f16, transposed
// so that a B-fragment = two contiguous 8-half runs per lane ----
constexpr int OFF_WIN = 0;                      // w_in^T  [64][32]  (K 16->pad 32)
constexpr int OFF_W0  = OFF_WIN + 64 * 32;      // W0^T    [256][64]
constexpr int OFF_W1  = OFF_W0 + 256 * 64;      // W1^T    [256][256]
constexpr int OFF_W2  = OFF_W1 + 256 * 256;     // W2^T    [256][256]
constexpr int OFF_AAT = OFF_W2 + 256 * 256;     // Aat^T   3 x [16][256]
constexpr int OFF_M1  = OFF_AAT + 3 * 16 * 256; // w1^T    [128][64]
constexpr int OFF_M2  = OFF_M1 + 128 * 64;      // w2^T    [256][128]
constexpr int WS_HALVES = OFF_M2 + 256 * 128;   // = 202752 halves (~396 KB)

// ---- per-wave LDS carve (bytes) ----
constexpr int SZ_H   = NN * HSTRIDE * 2;        // 8448  h (row major f16)
constexpr int SZ_WH  = NN * HSTRIDE * 2;        // 8448  wh (row major f16)
constexpr int SZ_WHT = F1 * WTSTRIDE * 2;       // 12288 wh^T (col major f16)
constexpr int SZ_A   = 4 * 16 * 16 * 2;         // 2048  alpha f16
constexpr int SZ_E   = 16 * 16 * 4;             // 1024  scores f32
constexpr int PER_WAVE_BYTES = SZ_H + SZ_WH + SZ_WHT + SZ_A + SZ_E; // 32256
constexpr int WAVES_PER_BLOCK = 8;
constexpr int SMEM_BYTES = PER_WAVE_BYTES * WAVES_PER_BLOCK;        // 258048

__device__ __forceinline__ void lds_fence() {
  asm volatile("s_wait_dscnt 0" ::: "memory");
}

__device__ __forceinline__ v8f wmma_f16(v16h a, v16h b, v8f c) {
  return __builtin_amdgcn_wmma_f32_16x16x32_f16(
      /*neg_a=*/false, a, /*neg_b=*/false, b,
      /*c_mod=*/(short)0, c, /*reuse_a=*/false, /*reuse_b=*/false);
}

// A fragment (16x32 f16): lane L holds row (L&15); halves 0..7 = K[off..off+7],
// halves 8..15 = K[off+16..off+23] with off = (L>=16 ? 8 : 0) (+kbase).
__device__ __forceinline__ v16h lds_afrag(const _Float16* p) {
  v8h lo = *(const v8h*)p;
  v8h hi = *(const v8h*)(p + 16);
  return __builtin_shufflevector(lo, hi, 0, 1, 2, 3, 4, 5, 6, 7,
                                 8, 9, 10, 11, 12, 13, 14, 15);
}

// 4-connected 4x4 grid adjacency + self loop, as a 16-bit neighbor mask.
__device__ __forceinline__ unsigned adjmask(int i) {
  int r = i >> 2, c = i & 3;
  unsigned m = 1u << i;
  if (c > 0) m |= 1u << (i - 1);
  if (c < 3) m |= 1u << (i + 1);
  if (r > 0) m |= 1u << (i - 4);
  if (r < 3) m |= 1u << (i + 4);
  return m;
}

// ---------------------------------------------------------------------------
// Prep: fp32 weights -> f16, transposed into WMMA-B friendly layout in d_ws.
// ---------------------------------------------------------------------------
__global__ void __launch_bounds__(256) gat_prep(
    const float* __restrict__ w_in, const float* __restrict__ W0,
    const float* __restrict__ as0, const float* __restrict__ ad0,
    const float* __restrict__ W1, const float* __restrict__ as1,
    const float* __restrict__ ad1, const float* __restrict__ W2,
    const float* __restrict__ as2, const float* __restrict__ ad2,
    const float* __restrict__ w1, const float* __restrict__ w2,
    _Float16* __restrict__ ws) {
  int idx = blockIdx.x * blockDim.x + threadIdx.x;
  if (idx >= WS_HALVES) return;
  float v = 0.0f;
  if (idx < OFF_W0) {                       // w_in^T [64][32], real K = 16
    int o = idx >> 5, k = idx & 31;
    v = (k < 16) ? w_in[k * 64 + o] : 0.0f;
  } else if (idx < OFF_W1) {                // W0^T [256][64]
    int t = idx - OFF_W0; int o = t >> 6, k = t & 63;
    v = W0[k * 256 + o];
  } else if (idx < OFF_W2) {                // W1^T [256][256]
    int t = idx - OFF_W1; int o = t >> 8, k = t & 255;
    v = W1[k * 256 + o];
  } else if (idx < OFF_AAT) {               // W2^T [256][256]
    int t = idx - OFF_W2; int o = t >> 8, k = t & 255;
    v = W2[k * 256 + o];
  } else if (idx < OFF_M1) {                // Aat^T: 3 x [16 cols][256 K]
    int t = idx - OFF_AAT; int l = t >> 12; int t2 = t & 4095;
    int col = t2 >> 8, k = t2 & 255;
    int h = k >> 6, c = k & 63;
    const float* as = (l == 0) ? as0 : (l == 1) ? as1 : as2;
    const float* ad = (l == 0) ? ad0 : (l == 1) ? ad1 : ad2;
    if (col == h)          v = as[h * 64 + c];   // cols 0..3  -> e_src head h
    else if (col == 4 + h) v = ad[h * 64 + c];   // cols 4..7  -> e_dst head h
    else                   v = 0.0f;             // cols 8..15 unused
  } else if (idx < OFF_M2) {                // w1^T [128][64]
    int t = idx - OFF_M1; int o = t >> 6, k = t & 63;
    v = w1[k * 128 + o];
  } else {                                  // w2^T [256][128]
    int t = idx - OFF_M2; int o = t >> 7, k = t & 127;
    v = w2[k * 256 + o];
  }
  ws[idx] = (_Float16)v;
}

// ---------------------------------------------------------------------------
// Main fused kernel: 8 waves/block, 1 graph/wave, all WMMA.
// ---------------------------------------------------------------------------
__global__ void __launch_bounds__(256, 1) gat_main(
    const float* __restrict__ x,    const float* __restrict__ b_in,
    const float* __restrict__ bb0,  const float* __restrict__ bb1,
    const float* __restrict__ bb2,  const float* __restrict__ b1,
    const float* __restrict__ ln_g, const float* __restrict__ ln_b,
    const float* __restrict__ b2,   const _Float16* __restrict__ ws,
    float* __restrict__ out, int nitems) {
  extern __shared__ char smem[];

  const int lane = threadIdx.x & 31;
  const int wave = threadIdx.x >> 5;
  const int item = blockIdx.x * WAVES_PER_BLOCK + wave;
  if (item >= nitems) return;  // whole-wave uniform

  const int n     = lane & 15;       // A row / B col / C col selector
  const int hi    = lane >> 4;       // lane half
  const int ksel8 = hi * 8;          // A: K sub-offset per ISA layout
  const int kh16  = hi * 16;         // prefetch K sub-offset (B frags use ksel8)
  const int mbase = hi * 8;          // C/D: row base per ISA layout

  char* wbase = smem + wave * PER_WAVE_BYTES;
  _Float16* hbuf  = (_Float16*)(wbase);                               // [16][264]
  _Float16* whbuf = (_Float16*)(wbase + SZ_H);                        // [16][264]
  _Float16* whT   = (_Float16*)(wbase + SZ_H + SZ_WH);                // [256][24]
  _Float16* abuf  = (_Float16*)(wbase + SZ_H + SZ_WH + SZ_WHT);       // [4][16][16]
  float*    ebuf  = (float*)   (wbase + SZ_H + SZ_WH + SZ_WHT + SZ_A);// [16][16]

  const _Float16* w_int = ws + OFF_WIN;
  const _Float16* const wT[3]   = {ws + OFF_W0, ws + OFF_W1, ws + OFF_W2};
  const _Float16* const aatT[3] = {ws + OFF_AAT, ws + OFF_AAT + 4096,
                                   ws + OFF_AAT + 8192};
  const float* const bias01[2] = {bb0, bb1};
  const int kin[3] = {64, 256, 256};

  // ===== stage 0: h = relu(X[16x16] @ w_in[16x64] + b_in) =====
  v16h a0;
  {
    const float* xb = x + (size_t)item * 256;  // x[b][c][node], c-major
    #pragma unroll
    for (int q = 0; q < 8; ++q) {
      a0[q]     = (_Float16)xb[(ksel8 + q) * 16 + n];  // K = input channel
      a0[q + 8] = (_Float16)0.0f;                      // K 16..31 zero pad
    }
  }
  #pragma unroll
  for (int nt = 0; nt < 4; ++nt) {
    int col = nt * 16 + n;
    v16h bfr = lds_afrag(w_int + col * 32 + ksel8);
    v8f acc = {};
    acc = wmma_f16(a0, bfr, acc);
    float bv = b_in[col];
    #pragma unroll
    for (int r = 0; r < 8; ++r) {
      float v = acc[r] + bv;
      hbuf[(mbase + r) * HSTRIDE + col] = (_Float16)(v > 0.0f ? v : 0.0f);
    }
  }
  lds_fence();

  // ===== 3 GAT layers =====
  for (int l = 0; l < 3; ++l) {
    const _Float16* W = wT[l];
    const int K  = kin[l];
    const int KT = K >> 5;

    // Preload A fragments of h (shared across the 16 N-tiles).
    v16h av[8];
    #pragma unroll
    for (int kt = 0; kt < 8; ++kt)
      if (kt < KT)
        av[kt] = lds_afrag(hbuf + n * HSTRIDE + kt * 32 + ksel8);

    // --- wh = h @ W  -> whbuf (row major) and whT (col major) ---
    for (int nt = 0; nt < 16; ++nt) {
      int col = nt * 16 + n;
      v8f acc = {};
      for (int kt = 0; kt < KT; ++kt) {
        if (kt + 1 < KT)  // stream next weight slice through L2
          __builtin_prefetch(W + col * K + (kt + 1) * 32 + kh16, 0, 0);
        v16h bfr = lds_afrag(W + col * K + kt * 32 + ksel8);
        acc = wmma_f16(av[kt], bfr, acc);
      }
      v8h pk;
      #pragma unroll
      for (int r = 0; r < 8; ++r) {
        _Float16 hv = (_Float16)acc[r];
        whbuf[(mbase + r) * HSTRIDE + col] = hv;
        pk[r] = hv;
      }
      *(v8h*)(whT + col * WTSTRIDE + mbase) = pk;  // wh^T[col][mbase..mbase+7]
    }
    lds_fence();

    // --- E = wh[16x256] @ Aat[256x16]: col h = e_src, col 4+h = e_dst ---
    {
      const _Float16* AT = aatT[l];
      v8f acc = {};
      #pragma unroll
      for (int kt = 0; kt < 8; ++kt) {
        v16h afr = lds_afrag(whbuf + n * HSTRIDE + kt * 32 + ksel8);
        v16h bfr = lds_afrag(AT + n * 256 + kt * 32 + ksel8);
        acc = wmma_f16(afr, bfr, acc);
      }
      #pragma unroll
      for (int r = 0; r < 8; ++r) ebuf[(mbase + r) * 16 + n] = acc[r];
    }
    lds_fence();

    // --- masked leaky-relu softmax: lane handles two (head, target) pairs ---
    #pragma unroll
    for (int pp = 0; pp < 2; ++pp) {
      int p = lane + pp * 32;
      int h = p >> 4, i = p & 15;
      float edst = ebuf[i * 16 + 4 + h];
      unsigned m = adjmask(i);
      float es[16], mx = -3.0e38f;
      #pragma unroll
      for (int j = 0; j < 16; ++j) {
        float e = edst + ebuf[j * 16 + h];
        e = (e > 0.0f) ? e : 0.2f * e;          // leaky relu, slope 0.2
        e = ((m >> j) & 1u) ? e : -1.0e9f;      // adjacency mask
        es[j] = e;
        mx = fmaxf(mx, e);
      }
      float s = 0.0f;
      #pragma unroll
      for (int j = 0; j < 16; ++j) { es[j] = __expf(es[j] - mx); s += es[j]; }
      float inv = 1.0f / s;
      #pragma unroll
      for (int j = 0; j < 16; ++j)
        abuf[(h * 16 + i) * 16 + j] = (_Float16)(es[j] * inv);
    }
    lds_fence();

    // --- aggregation: out = alpha_h[16x16] @ wh_h[16x64] per head -> hbuf ---
    v16h zfr;
    #pragma unroll
    for (int q = 0; q < 16; ++q) zfr[q] = (_Float16)0.0f;
    #pragma unroll
    for (int h = 0; h < 4; ++h) {
      // A = alpha (K = source node j, real 16, padded to 32)
      v16h afr = zfr;
      {
        v8h lo = *(const v8h*)(abuf + (h * 16 + n) * 16 + ksel8);
        #pragma unroll
        for (int q = 0; q < 8; ++q) afr[q] = lo[q];
      }
      #pragma unroll
      for (int nt = 0; nt < 4; ++nt) {
        int col = h * 64 + nt * 16 + n;
        // B = wh^T[col][j]: halves 0..7 carry K ksel8..ksel8+7, halves 8..15 zero pad
        v16h bfr = zfr;
        {
          v8h blo = *(const v8h*)(whT + col * WTSTRIDE + ksel8);
          #pragma unroll
          for (int q = 0; q < 8; ++q) bfr[q] = blo[q];
        }
        v8f acc = {};
        acc = wmma_f16(afr, bfr, acc);
        if (l < 2) {
          float bv = bias01[l][col];
          #pragma unroll
          for (int r = 0; r < 8; ++r) {
            float v = acc[r] + bv;
            hbuf[(mbase + r) * HSTRIDE + col] =
                (_Float16)(v > 0.0f ? v : 0.0f);  // + bias, relu
          }
        } else {
          #pragma unroll
          for (int r = 0; r < 8; ++r)
            hbuf[(mbase + r) * HSTRIDE + col] = (_Float16)acc[r];
        }
      }
    }
    lds_fence();
  }

  // ===== head-mean + bb2 + node-mean pool: g[64] =====
  float* gbuf = ebuf;           // reuse (64 floats)
  float* ybuf = (float*)abuf;   // reuse (128 floats)
  #pragma unroll
  for (int c0 = 0; c0 < 2; ++c0) {
    int c = lane + c0 * 32;
    float s = 0.0f;
    for (int i = 0; i < 16; ++i) {
      #pragma unroll
      for (int h = 0; h < 4; ++h) s += (float)hbuf[i * HSTRIDE + h * 64 + c];
    }
    gbuf[c] = s * (1.0f / 64.0f) + bb2[c];
  }
  lds_fence();

  // ===== y = LN(g @ w1 + b1), relu =====
  float yv[4], lsum = 0.0f, lsq = 0.0f;
  #pragma unroll
  for (int k = 0; k < 4; ++k) {
    int o = lane + k * 32;
    const _Float16* wr = ws + OFF_M1 + o * 64;
    float acc = b1[o];
    #pragma unroll 8
    for (int c = 0; c < 64; ++c) acc += gbuf[c] * (float)wr[c];
    yv[k] = acc; lsum += acc; lsq += acc * acc;
  }
  #pragma unroll
  for (int off = 16; off > 0; off >>= 1) {  // wave32 reduction
    lsum += __shfl_xor(lsum, off, 32);
    lsq  += __shfl_xor(lsq, off, 32);
  }
  float mu  = lsum * (1.0f / 128.0f);
  float var = lsq * (1.0f / 128.0f) - mu * mu;
  float inv = rsqrtf(var + 1e-5f);
  #pragma unroll
  for (int k = 0; k < 4; ++k) {
    int o = lane + k * 32;
    float t = (yv[k] - mu) * inv * ln_g[o] + ln_b[o];
    ybuf[o] = t > 0.0f ? t : 0.0f;
  }
  lds_fence();

  // ===== out = y @ w2 + b2 =====
  float* ob = out + (size_t)item * 256;
  #pragma unroll
  for (int k = 0; k < 8; ++k) {
    int o2 = lane + k * 32;
    const _Float16* wr = ws + OFF_M2 + o2 * 128;
    float acc = b2[o2];
    #pragma unroll 8
    for (int c = 0; c < 128; ++c) acc += ybuf[c] * (float)wr[c];
    ob[o2] = acc;
  }
}

// ---------------------------------------------------------------------------
extern "C" void kernel_launch(void* const* d_in, const int* in_sizes, int n_in,
                              void* d_out, int out_size, void* d_ws,
                              size_t ws_size, hipStream_t stream) {
  const float* x    = (const float*)d_in[0];
  const float* w_in = (const float*)d_in[1];
  const float* b_in = (const float*)d_in[2];
  const float* W0   = (const float*)d_in[3];
  const float* as0  = (const float*)d_in[4];
  const float* ad0  = (const float*)d_in[5];
  const float* bb0  = (const float*)d_in[6];
  const float* W1   = (const float*)d_in[7];
  const float* as1  = (const float*)d_in[8];
  const float* ad1  = (const float*)d_in[9];
  const float* bb1  = (const float*)d_in[10];
  const float* W2   = (const float*)d_in[11];
  const float* as2  = (const float*)d_in[12];
  const float* ad2  = (const float*)d_in[13];
  const float* bb2  = (const float*)d_in[14];
  const float* w1   = (const float*)d_in[15];
  const float* b1   = (const float*)d_in[16];
  const float* lng  = (const float*)d_in[17];
  const float* lnb  = (const float*)d_in[18];
  const float* w2   = (const float*)d_in[19];
  const float* b2   = (const float*)d_in[20];
  _Float16* ws = (_Float16*)d_ws;
  float* out = (float*)d_out;

  const int nitems = in_sizes[0] / 256;  // B = 16384

  // 252 KB dynamic LDS per block (WGP holds 320 KB) — raise the cap.
  (void)hipFuncSetAttribute((const void*)gat_main,
                            hipFuncAttributeMaxDynamicSharedMemorySize,
                            SMEM_BYTES);

  gat_prep<<<(WS_HALVES + 255) / 256, 256, 0, stream>>>(
      w_in, W0, as0, ad0, W1, as1, ad1, W2, as2, ad2, w1, w2, ws);

  const int blocks = (nitems + WAVES_PER_BLOCK - 1) / WAVES_PER_BLOCK;
  gat_main<<<blocks, 32 * WAVES_PER_BLOCK, SMEM_BYTES, stream>>>(
      x, b_in, bb0, bb1, bb2, b1, lng, lnb, b2, ws, out, nitems);
}
